// OSKAKERNEL_2310692405600
// MI455X (gfx1250) — matrix operand order repaired and checked by reading, not compiled, not run on hardware
//
#include <hip/hip_runtime.h>

typedef __attribute__((ext_vector_type(16))) _Float16 v16h;
typedef __attribute__((ext_vector_type(8)))  _Float16 h8;
typedef __attribute__((ext_vector_type(8)))  float    v8f;

#define WMMA_F32_F16(A, B, C) \
  __builtin_amdgcn_wmma_f32_16x16x32_f16(false, (A), false, (B), (short)0, (C), false, false)

__device__ __forceinline__ v16h pack16(h8 a, h8 b) {
  v16h r;
#pragma unroll
  for (int i = 0; i < 8; ++i) { r[i] = a[i]; r[i + 8] = b[i]; }
  return r;
}

// ---------------------------------------------------------------------------
// elementwise f32 -> f16 convert
// ---------------------------------------------------------------------------
__global__ __launch_bounds__(256) void cvt_f32_f16(const float* __restrict__ in,
                                                   _Float16* __restrict__ out, int n) {
  for (int i = blockIdx.x * blockDim.x + threadIdx.x; i < n; i += gridDim.x * blockDim.x)
    out[i] = (_Float16)in[i];
}

// ---------------------------------------------------------------------------
// C[M,N] f32 = A[M,K] f16 * B[N,K]^T f16   (both operands K-fast "NT" GEMM)
// block = 256 threads = 8 waves; tile 64(M) x 128(N); K-step 32.
// Double-buffered LDS tiles filled with global_load_async_to_lds_b128
// (ASYNCcnt path): prefetch tile i+1 while WMMAs consume tile i.
// ---------------------------------------------------------------------------
__global__ __launch_bounds__(256) void gemm_nt(const _Float16* __restrict__ A,
                                               const _Float16* __restrict__ B,
                                               float* __restrict__ C,
                                               int M, int N, int K) {
  __shared__ _Float16 lA[2][64][40];    // pitch 40 halves = 80B (16B-aligned rows)
  __shared__ _Float16 lB[2][128][40];
  const int tid = threadIdx.x;
  const int wave = tid >> 5, lane = tid & 31;
  const int mw = wave & 3, nw = wave >> 2;
  const int ln = lane & 15, hi = lane >> 4;
  const int m0 = blockIdx.y * 64, n0 = blockIdx.x * 128;

  // cooperative 16B chunk assignment: A = 256 chunks, B = 512 chunks
  const int rowA = tid >> 2, coA = (tid & 3) << 3;
  const int c2 = tid + 256;
  const int rowB1 = c2 >> 2, coB1 = (c2 & 3) << 3;

  const _Float16* gA  = &A[(size_t)(m0 + rowA) * K + coA];
  const _Float16* gB0 = &B[(size_t)(n0 + rowA) * K + coA];
  const _Float16* gB1 = &B[(size_t)(n0 + rowB1) * K + coB1];

  // issue 3 async global->LDS B128 transfers per thread for one K-tile
  auto issue = [&](int bufsel, int k0) {
    uint32_t dA  = (uint32_t)(uintptr_t)&lA[bufsel][rowA][coA];
    uint32_t dB0 = (uint32_t)(uintptr_t)&lB[bufsel][rowA][coA];
    uint32_t dB1 = (uint32_t)(uintptr_t)&lB[bufsel][rowB1][coB1];
    asm volatile("global_load_async_to_lds_b128 %0, %1, off"
                 :: "v"(dA),  "v"(gA  + k0) : "memory");
    asm volatile("global_load_async_to_lds_b128 %0, %1, off"
                 :: "v"(dB0), "v"(gB0 + k0) : "memory");
    asm volatile("global_load_async_to_lds_b128 %0, %1, off"
                 :: "v"(dB1), "v"(gB1 + k0) : "memory");
  };

  v8f acc[4] = {};
  issue(0, 0);
  int buf = 0;
  for (int k0 = 0; k0 < K; k0 += 32, buf ^= 1) {
    const int nxt = k0 + 32;
    if (nxt < K) {
      issue(buf ^ 1, nxt);
      asm volatile("s_wait_asynccnt 0x3" ::: "memory");  // drain current tile's 3
    } else {
      asm volatile("s_wait_asynccnt 0x0" ::: "memory");
    }
    __syncthreads();   // all waves' async fills for tile `buf` are visible

    // A fragment: lane holds row m; halves p0..7 -> K=hi*8+p, p8..15 -> K=16+hi*8+p
    const int m = mw * 16 + ln;
    h8 a0 = *(const h8*)&lA[buf][m][hi * 8];
    h8 a1 = *(const h8*)&lA[buf][m][16 + hi * 8];
    v16h af = pack16(a0, a1);

#pragma unroll
    for (int t = 0; t < 4; ++t) {
      // B fragment: lane holds column n; halves p0..7 -> K=hi*8+p, p8..15 -> K=16+hi*8+p
      const int n = nw * 64 + t * 16 + ln;
      h8 b0 = *(const h8*)&lB[buf][n][hi * 8];
      h8 b1 = *(const h8*)&lB[buf][n][16 + hi * 8];
      v16h bf = pack16(b0, b1);
      acc[t] = WMMA_F32_F16(af, bf, acc[t]);
    }
    __syncthreads();   // all waves done reading `buf` before it is refilled
  }

  // C layout: VGPR r, lane -> (M = r + hi*8, N = ln)
#pragma unroll
  for (int t = 0; t < 4; ++t)
#pragma unroll
    for (int r = 0; r < 8; ++r) {
      const int row = m0 + mw * 16 + hi * 8 + r;
      const int col = n0 + nw * 64 + t * 16 + ln;
      C[(size_t)row * N + col] = acc[t][r];
    }
}

// ---------------------------------------------------------------------------
// per-(b,s,h) l2-normalize a 64-vector, emit f16 into [B*H, S, 64]
// one wave per group; 2 elements per lane; shuffle reduction (wave32)
// ---------------------------------------------------------------------------
__global__ __launch_bounds__(256) void pack_qk(const float* __restrict__ in,
                                               _Float16* __restrict__ out) {
  const int g = blockIdx.x * 8 + (threadIdx.x >> 5);
  const int lane = threadIdx.x & 31;
  const int h = g & 15, bs = g >> 4;           // bs = b*1024 + s
  const int s = bs & 1023, b = bs >> 10;
  const float* p = in + (size_t)bs * 1024 + h * 64;
  float x0 = p[lane * 2], x1 = p[lane * 2 + 1];
  float ss = x0 * x0 + x1 * x1;
#pragma unroll
  for (int o = 16; o > 0; o >>= 1) ss += __shfl_xor(ss, o, 32);
  const float inv = 1.0f / fmaxf(sqrtf(ss), 1e-12f);
  _Float16* q = out + ((size_t)(b * 16 + h) * 1024 + s) * 64;
  q[lane * 2]     = (_Float16)(x0 * inv);
  q[lane * 2 + 1] = (_Float16)(x1 * inv);
}

// ---------------------------------------------------------------------------
// V: f32 [B,S,D] -> f16 TRANSPOSED [B*H, 64, S] (so attn B-frags are contiguous)
// ---------------------------------------------------------------------------
__global__ __launch_bounds__(256) void pack_v(const float* __restrict__ in,
                                              _Float16* __restrict__ out) {
  const int g = blockIdx.x * 8 + (threadIdx.x >> 5);
  const int lane = threadIdx.x & 31;
  const int h = g & 15, bs = g >> 4;
  const int s = bs & 1023, b = bs >> 10;
  const float* p = in + (size_t)bs * 1024 + h * 64;
  float x0 = p[lane * 2], x1 = p[lane * 2 + 1];
  _Float16* q = out + (size_t)(b * 16 + h) * 64 * 1024 + s;
  q[(size_t)(lane * 2) * 1024]     = (_Float16)x0;
  q[(size_t)(lane * 2 + 1) * 1024] = (_Float16)x1;
}

// ---------------------------------------------------------------------------
// Gegenbauer causal attention. grid = (S/128, B*H); 8 waves, wave owns 16 q-rows.
// Per 32-k block: 4 WMMA -> sim(16x32); per-lane 3-term recurrence + causal mask;
// relayout C->A through per-wave LDS tile; 4 WMMA into 16x64 f32 accumulators.
// ---------------------------------------------------------------------------
__global__ __launch_bounds__(256) void attn_kernel(const _Float16* __restrict__ Q,
                                                   const _Float16* __restrict__ Kh,
                                                   const _Float16* __restrict__ Vt,
                                                   const float* __restrict__ poly,
                                                   const float* __restrict__ degs,
                                                   float* __restrict__ out) {
  constexpr int S = 1024;
  constexpr float LAM = 31.5f;                 // (Q_DIM-1)/2
  __shared__ _Float16 wt[8][16][40];           // per-wave 16x32 weight tile, padded pitch
  const int bh = blockIdx.y, h = bh & 15, b = bh >> 4;
  const int tid = threadIdx.x, wave = tid >> 5, lane = tid & 31;
  const int ln = lane & 15, hi = lane >> 4;
  const int m0 = blockIdx.x * 128 + wave * 16;

  float wg[9];
  {
    const float d = degs[h];
#pragma unroll
    for (int kk = 0; kk < 9; ++kk) {
      const float gte = fminf(fmaxf(d - (float)kk + 1.0f, 0.0f), 1.0f);
      wg[kk] = poly[h * 9 + kk] * gte;
    }
  }

  const _Float16* Qb = Q  + (size_t)bh * S * 64;
  const _Float16* Kb = Kh + (size_t)bh * S * 64;
  const _Float16* Vb = Vt + (size_t)bh * 64 * S;

  // resident Q A-fragments covering dh=64 (two K=32 fragments)
  v16h qa[2];
#pragma unroll
  for (int sf = 0; sf < 2; ++sf) {
    const _Float16* qp = Qb + (size_t)(m0 + ln) * 64 + sf * 32;
    qa[sf] = pack16(*(const h8*)(qp + hi * 8), *(const h8*)(qp + 16 + hi * 8));
  }

  v8f oacc[4] = {};
  const int kend = m0 + 16;                    // causal: max k index needed is m0+15

  for (int k0 = 0; k0 < kend; k0 += 32) {
#pragma unroll
    for (int sub = 0; sub < 2; ++sub) {
      const int kt = k0 + sub * 16;
      v8f sim = {};
      if (kt < kend) {
#pragma unroll
        for (int sf = 0; sf < 2; ++sf) {
          // B frag of K^T: lane = k-row n, halves -> dh index c = sf*32 + hi*8 + p (p<8), sf*32 + 16 + hi*8 + (p-8) (p>=8)
          const _Float16* kp = Kb + (size_t)(kt + ln) * 64 + sf * 32 + hi * 8;
          v16h bf = pack16(*(const h8*)kp, *(const h8*)(kp + 16));
          sim = WMMA_F32_F16(qa[sf], bf, sim);
        }
      }
      // Gegenbauer recurrence + causal mask, write f16 weights to per-wave LDS tile
#pragma unroll
      for (int r = 0; r < 8; ++r) {
        const int mg = m0 + hi * 8 + r;
        const int ng = kt + ln;
        float phi = 0.0f;
        if (ng <= mg) {
          const float sv = sim[r];
          float Rp = 1.0f, Rc = sv;
          phi = wg[0] + wg[1] * sv;
#pragma unroll
          for (int kk = 2; kk <= 8; ++kk) {
            const float c1 = 2.0f * ((float)kk + LAM - 1.0f) / ((float)kk + 2.0f * LAM - 1.0f);
            const float c2 = ((float)kk - 1.0f) / ((float)kk + 2.0f * LAM - 1.0f);
            const float Rn = c1 * sv * Rc - c2 * Rp;
            phi += wg[kk] * Rn;
            Rp = Rc; Rc = Rn;
          }
        }
        wt[wave][hi * 8 + r][sub * 16 + ln] = (_Float16)phi;
      }
    }
    // per-wave LDS is in order on CDNA5; fence the compiler + drain DScnt
    asm volatile("s_wait_dscnt 0" ::: "memory");

    // weights A-fragment (16x32) from LDS
    v16h wf = pack16(*(const h8*)&wt[wave][ln][hi * 8],
                     *(const h8*)&wt[wave][ln][16 + hi * 8]);

#pragma unroll
    for (int t = 0; t < 4; ++t) {
      // V^T B frag: lane = dh column, halves -> k = k0 + hi*8 + p (p<8), k0 + 16 + hi*8 + (p-8) (p>=8)
      const _Float16* vp = Vb + (size_t)(t * 16 + ln) * S + k0 + hi * 8;
      v16h bf = pack16(*(const h8*)vp, *(const h8*)(vp + 16));
      oacc[t] = WMMA_F32_F16(wf, bf, oacc[t]);
    }
    asm volatile("" ::: "memory");   // keep next iter's LDS writes after these reads
  }

  // epilogue: divide by (q_index+1), write f32 [B,S,D] with D = h*64 + d
#pragma unroll
  for (int t = 0; t < 4; ++t)
#pragma unroll
    for (int r = 0; r < 8; ++r) {
      const int mg = m0 + hi * 8 + r;
      const float val = oacc[t][r] / (float)(mg + 1);
      out[((size_t)b * S + mg) * 1024 + h * 64 + t * 16 + ln] = val;
    }
}

// ---------------------------------------------------------------------------
// RMSNorm over each 64-wide head row, scale by norm_w, emit f16 [B,S,D]
// ---------------------------------------------------------------------------
__global__ __launch_bounds__(256) void rms_pack(const float* __restrict__ in,
                                                const float* __restrict__ nw,
                                                _Float16* __restrict__ out) {
  const int g = blockIdx.x * 8 + (threadIdx.x >> 5);
  const int lane = threadIdx.x & 31;
  const int h = g & 15, bs = g >> 4;
  const float* p = in + (size_t)bs * 1024 + h * 64;
  float x0 = p[lane * 2], x1 = p[lane * 2 + 1];
  float ss = x0 * x0 + x1 * x1;
#pragma unroll
  for (int o = 16; o > 0; o >>= 1) ss += __shfl_xor(ss, o, 32);
  const float inv = rsqrtf(ss * (1.0f / 64.0f) + 1e-6f);
  _Float16* q = out + (size_t)bs * 1024 + h * 64;
  q[lane * 2]     = (_Float16)(x0 * inv * nw[lane * 2]);
  q[lane * 2 + 1] = (_Float16)(x1 * inv * nw[lane * 2 + 1]);
}

// ---------------------------------------------------------------------------
extern "C" void kernel_launch(void* const* d_in, const int* in_sizes, int n_in,
                              void* d_out, int out_size, void* d_ws, size_t ws_size,
                              hipStream_t stream) {
  (void)in_sizes; (void)n_in; (void)out_size; (void)ws_size;
  const float* x    = (const float*)d_in[0];
  const float* Wq   = (const float*)d_in[1];
  const float* Wk   = (const float*)d_in[2];
  const float* Wv   = (const float*)d_in[3];
  const float* Wo   = (const float*)d_in[4];
  const float* poly = (const float*)d_in[5];
  const float* nw   = (const float*)d_in[6];
  const float* degs = (const float*)d_in[7];

  char* ws = (char*)d_ws;
  size_t off = 0;
  auto take = [&](size_t bytes) -> char* {
    char* p = ws + off;
    off += (bytes + 255) & ~(size_t)255;
    return p;
  };
  const size_t MSD = 2048ull * 1024ull;   // B*S*D elements
  const size_t WSZ = 1024ull * 1024ull;   // D*D elements

  _Float16* xh  = (_Float16*)take(MSD * 2);
  _Float16* wqh = (_Float16*)take(WSZ * 2);
  _Float16* wkh = (_Float16*)take(WSZ * 2);
  _Float16* wvh = (_Float16*)take(WSZ * 2);
  _Float16* woh = (_Float16*)take(WSZ * 2);
  float*    pf  = (float*)   take(MSD * 4);   // projection scratch; reused as attn out
  _Float16* qh  = (_Float16*)take(MSD * 2);   // [B*H, S, 64] normalized Q
  _Float16* kh  = (_Float16*)take(MSD * 2);   // [B*H, S, 64] normalized K
  _Float16* vt  = (_Float16*)take(MSD * 2);   // [B*H, 64, S] transposed V
  _Float16* ah  = (_Float16*)take(MSD * 2);   // post-RMS heads, f16 [B,S,D]

  cvt_f32_f16<<<512, 256, 0, stream>>>(x,  xh,  (int)MSD);
  cvt_f32_f16<<<256, 256, 0, stream>>>(Wq, wqh, (int)WSZ);
  cvt_f32_f16<<<256, 256, 0, stream>>>(Wk, wkh, (int)WSZ);
  cvt_f32_f16<<<256, 256, 0, stream>>>(Wv, wvh, (int)WSZ);
  cvt_f32_f16<<<256, 256, 0, stream>>>(Wo, woh, (int)WSZ);

  const dim3 gg(1024 / 128, 2048 / 64);   // (N tiles, M tiles)
  gemm_nt<<<gg, 256, 0, stream>>>(xh, wqh, pf, 2048, 1024, 1024);
  pack_qk<<<4096, 256, 0, stream>>>(pf, qh);
  gemm_nt<<<gg, 256, 0, stream>>>(xh, wkh, pf, 2048, 1024, 1024);
  pack_qk<<<4096, 256, 0, stream>>>(pf, kh);
  gemm_nt<<<gg, 256, 0, stream>>>(xh, wvh, pf, 2048, 1024, 1024);
  pack_v<<<4096, 256, 0, stream>>>(pf, vt);

  attn_kernel<<<dim3(1024 / 128, 32), 256, 0, stream>>>(qh, kh, vt, poly, degs, pf);
  rms_pack<<<4096, 256, 0, stream>>>(pf, nw, ah);

  gemm_nt<<<gg, 256, 0, stream>>>(ah, woh, (float*)d_out, 2048, 1024, 1024);
}
